// GroupedExperts_70136815943759
// MI455X (gfx1250) — compile-verified, matrix operand order checked by reading, not run on hardware
//
#include <hip/hip_runtime.h>

// ---------------- problem constants ----------------
#define T_TOK   8192
#define DMODEL  4096
#define HDIM    1024
#define NEXP    8
#define NRANK   2
#define ALIGN_M 16
#define MAX_LEN (T_TOK + NEXP * ALIGN_M)          // 8320
#define MAX_TILES (NEXP + (MAX_LEN + 127) / 128)  // 73

// ---------------- types ----------------
typedef __attribute__((ext_vector_type(16))) __bf16 v16bf;
typedef __attribute__((ext_vector_type(8)))  float  v8f;
typedef int v4i_g __attribute__((__vector_size__(16)));   // builtin's pointee type

union Frag {            // 8 VGPRs = 16 bf16 = two 16B chunks
    uint4 u[2];
    v16bf v;
};
union F4 { float4 v; float f[4]; };

__device__ inline unsigned short f2bf(float f) {  // round-to-nearest-even
    unsigned int u = __float_as_uint(f);
    u += 0x7FFFu + ((u >> 16) & 1u);
    return (unsigned short)(u >> 16);
}
__device__ inline unsigned int pack2(float lo, float hi) {
    return (unsigned int)f2bf(lo) | ((unsigned int)f2bf(hi) << 16);
}
__device__ inline v8f wmma_bf16(const Frag& a, const Frag& b, v8f c) {
    return __builtin_amdgcn_wmma_f32_16x16x32_bf16(
        false, a.v, false, b.v, (short)0, c, false, false);
}

// ---- CDNA5 async global->LDS copy (ASYNCcnt-tracked), 16B per lane ----
__device__ inline void async_copy16(const void* g, void* l) {
#if __has_builtin(__builtin_amdgcn_global_load_async_to_lds_b128)
    __builtin_amdgcn_global_load_async_to_lds_b128((v4i_g*)g, (v4i_g*)l, 0, 0);
#else
    unsigned laddr = (unsigned)(unsigned long)(__attribute__((address_space(3))) void*)l;
    asm volatile("global_load_async_to_lds_b128 %0, %1, off"
                 :: "v"(laddr), "v"(g) : "memory");
#endif
}
__device__ inline void wait_async_le4() {
#if __has_builtin(__builtin_amdgcn_s_wait_asynccnt)
    __builtin_amdgcn_s_wait_asynccnt(4);
#else
    asm volatile("s_wait_asynccnt 0x4" ::: "memory");
#endif
}
__device__ inline void wait_async_0() {
#if __has_builtin(__builtin_amdgcn_s_wait_asynccnt)
    __builtin_amdgcn_s_wait_asynccnt(0);
#else
    asm volatile("s_wait_asynccnt 0x0" ::: "memory");
#endif
}

// ---------------- K1: permutation + tile table ----------------
__global__ __launch_bounds__(256) void setup_kernel(
    const int* __restrict__ counts, int* __restrict__ hdr,
    int* __restrict__ perm, int* __restrict__ tile_e,
    int* __restrict__ tile_m0, int* __restrict__ tile_rows)
{
    __shared__ int s_ws[NEXP * NRANK], s_len[NEXP * NRANK], s_src[NEXP * NRANK];
    if (threadIdx.x == 0) {
        int cnt[NRANK * NEXP], start[NRANK * NEXP];
        int acc = 0;
        for (int i = 0; i < NRANK * NEXP; ++i) { cnt[i] = counts[i]; }
        for (int i = 0; i < NRANK * NEXP; ++i) { start[i] = acc; acc += cnt[i]; }
        int msz[NEXP], moff[NEXP]; int mo = 0;
        for (int e = 0; e < NEXP; ++e) {
            int tot = 0;
            for (int r = 0; r < NRANK; ++r) tot += cnt[r * NEXP + e];
            int ms = tot < ALIGN_M ? ALIGN_M : tot;
            ms = (ms + ALIGN_M - 1) / ALIGN_M * ALIGN_M;
            msz[e] = ms; moff[e] = mo; mo += ms;
        }
        for (int e = 0; e < NEXP; ++e) {          // expert-major, rank-minor segs
            int wo = moff[e];
            for (int r = 0; r < NRANK; ++r) {
                int s = e * NRANK + r;
                s_ws[s] = wo; s_len[s] = cnt[r * NEXP + e]; s_src[s] = start[r * NEXP + e];
                wo += cnt[r * NEXP + e];
            }
        }
        int t = 0;                                 // per-expert 128-row tiles
        for (int e = 0; e < NEXP; ++e)
            for (int lo = 0; lo < msz[e]; lo += 128) {
                tile_e[t] = e; tile_m0[t] = moff[e] + lo;
                int rr = msz[e] - lo; tile_rows[t] = rr < 128 ? rr : 128;
                ++t;
            }
        hdr[0] = t; hdr[1] = mo;
    }
    __syncthreads();
    for (int pos = threadIdx.x; pos < MAX_LEN; pos += 256) {
        int idx = 0;
        #pragma unroll
        for (int j = 1; j < NEXP * NRANK; ++j) if (pos >= s_ws[j]) idx = j;
        int within = pos - s_ws[idx];
        perm[pos] = (within < s_len[idx]) ? (s_src[idx] + within) : T_TOK;
    }
}

// ---------------- K2: gather x[perm] -> bf16 xp ----------------
__global__ __launch_bounds__(256) void gather_kernel(
    const float* __restrict__ x, const int* __restrict__ perm,
    unsigned short* __restrict__ xp)
{
    size_t p = ((size_t)blockIdx.x * 256 + threadIdx.x) * 8;
    int row = (int)(p / DMODEL);
    int col = (int)(p % DMODEL);
    int src = perm[row];
    uint4 o;
    if (src < T_TOK) {
        const float4 v0 = *(const float4*)(x + (size_t)src * DMODEL + col);
        const float4 v1 = *(const float4*)(x + (size_t)src * DMODEL + col + 4);
        o.x = pack2(v0.x, v0.y); o.y = pack2(v0.z, v0.w);
        o.z = pack2(v1.x, v1.y); o.w = pack2(v1.z, v1.w);
    } else {
        o = make_uint4(0u, 0u, 0u, 0u);
    }
    *(uint4*)(xp + p) = o;
}

// ---------------- K3: h = silu(xp@w1) * (xp@w3)  (bf16 WMMA, fused) -------
// block tile 128(M) x 64(N), K-chunk 64, 8 waves (4x2), wave tile 32x32.
// A double-buffered via async global->LDS; B prefetched into registers.
__global__ __launch_bounds__(256) void gemm13_kernel(
    const unsigned short* __restrict__ xp, const float* __restrict__ w1,
    const float* __restrict__ w3, unsigned short* __restrict__ hbuf,
    const int* __restrict__ hdr, const int* __restrict__ tile_e,
    const int* __restrict__ tile_m0, const int* __restrict__ tile_rows)
{
    __shared__ unsigned short As[2][128 * 80];   // [m][k], 160B rows
    __shared__ unsigned short B1s[64 * 80];      // transposed: [n][k]
    __shared__ unsigned short B3s[64 * 80];

    const int n_tiles = hdr[0];
    const int t = blockIdx.x;
    if (t >= n_tiles) return;
    const int e = tile_e[t], m0 = tile_m0[t], rows = tile_rows[t];
    const int n0 = blockIdx.y * 64;
    const float* W1 = w1 + (size_t)e * DMODEL * HDIM;
    const float* W3 = w3 + (size_t)e * DMODEL * HDIM;

    const int tid  = threadIdx.x;
    const int lane = tid & 31, wave = tid >> 5;
    const int wm = wave >> 1, wn = wave & 1;
    const int lr = lane & 15, hf = lane >> 4;

    // per-thread B item coords: 2 items of (2 adjacent k) x (4 adjacent n)
    int kI[2], ngI[2];
    #pragma unroll
    for (int i = 0; i < 2; ++i) {
        int id = tid + i * 256;
        kI[i]  = (id >> 4) * 2;
        ngI[i] = (id & 15) * 4;
    }
    // per-thread A async-copy coords: 4 x 16B
    int arI[4], aqI[4];
    #pragma unroll
    for (int i = 0; i < 4; ++i) {
        int id = tid + i * 256;
        arI[i] = id >> 3; aqI[i] = id & 7;
    }

    v8f c1[2][2], c3[2][2];
    #pragma unroll
    for (int i = 0; i < 2; ++i)
        #pragma unroll
        for (int j = 0; j < 2; ++j)
            #pragma unroll
            for (int k = 0; k < 8; ++k) { c1[i][j][k] = 0.f; c3[i][j][k] = 0.f; }

    F4 rb1[2][2], rb3[2][2];

    // ---- prologue: async A(chunk 0), B(chunk 0) into regs ----
    #pragma unroll
    for (int i = 0; i < 4; ++i) {
        int row = m0 + arI[i]; if (row > MAX_LEN - 1) row = MAX_LEN - 1;
        async_copy16(xp + (size_t)row * DMODEL + aqI[i] * 8,
                     &As[0][arI[i] * 80 + aqI[i] * 8]);
    }
    #pragma unroll
    for (int i = 0; i < 2; ++i) {
        const float* p1 = W1 + (size_t)kI[i] * HDIM + n0 + ngI[i];
        const float* p3 = W3 + (size_t)kI[i] * HDIM + n0 + ngI[i];
        rb1[i][0].v = *(const float4*)p1; rb1[i][1].v = *(const float4*)(p1 + HDIM);
        rb3[i][0].v = *(const float4*)p3; rb3[i][1].v = *(const float4*)(p3 + HDIM);
    }

    const int NCH = DMODEL / 64;
    for (int i = 0; i < NCH; ++i) {
        unsigned short* Acur = As[i & 1];
        unsigned short* Anxt = As[(i + 1) & 1];
        const int k0n = (i + 1) * 64;
        // store B regs (chunk i) -> LDS, packed pairs along k
        #pragma unroll
        for (int q = 0; q < 2; ++q)
            #pragma unroll
            for (int j = 0; j < 4; ++j) {
                *(unsigned int*)&B1s[(ngI[q] + j) * 80 + kI[q]] =
                    pack2(rb1[q][0].f[j], rb1[q][1].f[j]);
                *(unsigned int*)&B3s[(ngI[q] + j) * 80 + kI[q]] =
                    pack2(rb3[q][0].f[j], rb3[q][1].f[j]);
            }
        if (i + 1 < NCH) {
            // async A(chunk i+1) into alternate buffer; B(chunk i+1) into regs
            #pragma unroll
            for (int q = 0; q < 4; ++q) {
                int row = m0 + arI[q]; if (row > MAX_LEN - 1) row = MAX_LEN - 1;
                async_copy16(xp + (size_t)row * DMODEL + k0n + aqI[q] * 8,
                             &Anxt[arI[q] * 80 + aqI[q] * 8]);
            }
            #pragma unroll
            for (int q = 0; q < 2; ++q) {
                const float* p1 = W1 + (size_t)(k0n + kI[q]) * HDIM + n0 + ngI[q];
                const float* p3 = W3 + (size_t)(k0n + kI[q]) * HDIM + n0 + ngI[q];
                rb1[q][0].v = *(const float4*)p1; rb1[q][1].v = *(const float4*)(p1 + HDIM);
                rb3[q][0].v = *(const float4*)p3; rb3[q][1].v = *(const float4*)(p3 + HDIM);
            }
            wait_async_le4();      // chunk i landed; chunk i+1 in flight
        } else {
            wait_async_0();
        }
        __syncthreads();
        // ---- compute chunk i ----
        #pragma unroll
        for (int kk = 0; kk < 64; kk += 32) {
            Frag a[2], b1[2], b3[2];
            #pragma unroll
            for (int tm = 0; tm < 2; ++tm) {
                int m = wm * 32 + tm * 16 + lr;
                a[tm].u[0] = *(const uint4*)(Acur + m * 80 + kk + hf * 8);
                a[tm].u[1] = *(const uint4*)(Acur + m * 80 + kk + hf * 8 + 16);
            }
            #pragma unroll
            for (int tn = 0; tn < 2; ++tn) {
                int n = wn * 32 + tn * 16 + lr;
                b1[tn].u[0] = *(const uint4*)(B1s + n * 80 + kk + hf * 8);
                b1[tn].u[1] = *(const uint4*)(B1s + n * 80 + kk + hf * 8 + 16);
                b3[tn].u[0] = *(const uint4*)(B3s + n * 80 + kk + hf * 8);
                b3[tn].u[1] = *(const uint4*)(B3s + n * 80 + kk + hf * 8 + 16);
            }
            #pragma unroll
            for (int tm = 0; tm < 2; ++tm)
                #pragma unroll
                for (int tn = 0; tn < 2; ++tn) {
                    c1[tm][tn] = wmma_bf16(a[tm], b1[tn], c1[tm][tn]);
                    c3[tm][tn] = wmma_bf16(a[tm], b3[tn], c3[tm][tn]);
                }
        }
        __syncthreads();
    }
    // epilogue: SwiGLU, store h as bf16
    #pragma unroll
    for (int tm = 0; tm < 2; ++tm)
        #pragma unroll
        for (int tn = 0; tn < 2; ++tn)
            #pragma unroll
            for (int j = 0; j < 8; ++j) {
                int m = wm * 32 + tm * 16 + hf * 8 + j;
                if (m < rows) {
                    float a = c1[tm][tn][j];
                    float g = c3[tm][tn][j];
                    float val = (a / (1.f + __expf(-a))) * g;
                    int col = n0 + wn * 32 + tn * 16 + lr;
                    hbuf[(size_t)(m0 + m) * HDIM + col] = f2bf(val);
                }
            }
}

// ---------------- K4: out[perm] = h @ w2 (bf16 WMMA + scatter) ----------
__global__ __launch_bounds__(256) void gemm2_kernel(
    const unsigned short* __restrict__ hbuf, const float* __restrict__ w2,
    const int* __restrict__ perm, float* __restrict__ out,
    const int* __restrict__ hdr, const int* __restrict__ tile_e,
    const int* __restrict__ tile_m0, const int* __restrict__ tile_rows)
{
    __shared__ unsigned short As[2][128 * 80];
    __shared__ unsigned short Bs[64 * 80];

    const int n_tiles = hdr[0];
    const int t = blockIdx.x;
    if (t >= n_tiles) return;
    const int e = tile_e[t], m0 = tile_m0[t], rows = tile_rows[t];
    const int n0 = blockIdx.y * 64;
    const float* W2 = w2 + (size_t)e * HDIM * DMODEL;

    const int tid  = threadIdx.x;
    const int lane = tid & 31, wave = tid >> 5;
    const int wm = wave >> 1, wn = wave & 1;
    const int lr = lane & 15, hf = lane >> 4;

    int kI[2], ngI[2];
    #pragma unroll
    for (int i = 0; i < 2; ++i) {
        int id = tid + i * 256;
        kI[i]  = (id >> 4) * 2;
        ngI[i] = (id & 15) * 4;
    }
    int arI[4], aqI[4];
    #pragma unroll
    for (int i = 0; i < 4; ++i) {
        int id = tid + i * 256;
        arI[i] = id >> 3; aqI[i] = id & 7;
    }

    v8f c[2][2];
    #pragma unroll
    for (int i = 0; i < 2; ++i)
        #pragma unroll
        for (int j = 0; j < 2; ++j)
            #pragma unroll
            for (int k = 0; k < 8; ++k) c[i][j][k] = 0.f;

    F4 rb[2][2];

    #pragma unroll
    for (int i = 0; i < 4; ++i) {
        int row = m0 + arI[i]; if (row > MAX_LEN - 1) row = MAX_LEN - 1;
        async_copy16(hbuf + (size_t)row * HDIM + aqI[i] * 8,
                     &As[0][arI[i] * 80 + aqI[i] * 8]);
    }
    #pragma unroll
    for (int i = 0; i < 2; ++i) {
        const float* p2 = W2 + (size_t)kI[i] * DMODEL + n0 + ngI[i];
        rb[i][0].v = *(const float4*)p2; rb[i][1].v = *(const float4*)(p2 + DMODEL);
    }

    const int NCH = HDIM / 64;
    for (int i = 0; i < NCH; ++i) {
        unsigned short* Acur = As[i & 1];
        unsigned short* Anxt = As[(i + 1) & 1];
        const int k0n = (i + 1) * 64;
        #pragma unroll
        for (int q = 0; q < 2; ++q)
            #pragma unroll
            for (int j = 0; j < 4; ++j)
                *(unsigned int*)&Bs[(ngI[q] + j) * 80 + kI[q]] =
                    pack2(rb[q][0].f[j], rb[q][1].f[j]);
        if (i + 1 < NCH) {
            #pragma unroll
            for (int q = 0; q < 4; ++q) {
                int row = m0 + arI[q]; if (row > MAX_LEN - 1) row = MAX_LEN - 1;
                async_copy16(hbuf + (size_t)row * HDIM + k0n + aqI[q] * 8,
                             &Anxt[arI[q] * 80 + aqI[q] * 8]);
            }
            #pragma unroll
            for (int q = 0; q < 2; ++q) {
                const float* p2 = W2 + (size_t)(k0n + kI[q]) * DMODEL + n0 + ngI[q];
                rb[q][0].v = *(const float4*)p2; rb[q][1].v = *(const float4*)(p2 + DMODEL);
            }
            wait_async_le4();
        } else {
            wait_async_0();
        }
        __syncthreads();
        #pragma unroll
        for (int kk = 0; kk < 64; kk += 32) {
            Frag a[2], b[2];
            #pragma unroll
            for (int tm = 0; tm < 2; ++tm) {
                int m = wm * 32 + tm * 16 + lr;
                a[tm].u[0] = *(const uint4*)(Acur + m * 80 + kk + hf * 8);
                a[tm].u[1] = *(const uint4*)(Acur + m * 80 + kk + hf * 8 + 16);
            }
            #pragma unroll
            for (int tn = 0; tn < 2; ++tn) {
                int n = wn * 32 + tn * 16 + lr;
                b[tn].u[0] = *(const uint4*)(Bs + n * 80 + kk + hf * 8);
                b[tn].u[1] = *(const uint4*)(Bs + n * 80 + kk + hf * 8 + 16);
            }
            #pragma unroll
            for (int tm = 0; tm < 2; ++tm)
                #pragma unroll
                for (int tn = 0; tn < 2; ++tn)
                    c[tm][tn] = wmma_bf16(a[tm], b[tn], c[tm][tn]);
        }
        __syncthreads();
    }
    // epilogue: scatter rows through perm into zero-initialized out
    #pragma unroll
    for (int tm = 0; tm < 2; ++tm)
        #pragma unroll
        for (int tn = 0; tn < 2; ++tn)
            #pragma unroll
            for (int j = 0; j < 8; ++j) {
                int m = wm * 32 + tm * 16 + hf * 8 + j;
                if (m < rows) {
                    int dr = perm[m0 + m];
                    if (dr < T_TOK) {
                        int col = n0 + wn * 32 + tn * 16 + lr;
                        out[(size_t)dr * DMODEL + col] = c[tm][tn][j];
                    }
                }
            }
}

// ---------------- launcher ----------------
extern "C" void kernel_launch(void* const* d_in, const int* in_sizes, int n_in,
                              void* d_out, int out_size, void* d_ws, size_t ws_size,
                              hipStream_t stream) {
    const float* x      = (const float*)d_in[0];
    const float* w1     = (const float*)d_in[1];
    const float* w2     = (const float*)d_in[2];
    const float* w3     = (const float*)d_in[3];
    const int*   counts = (const int*)d_in[4];
    float* out = (float*)d_out;

    char* ws = (char*)d_ws;
    int* hdr       = (int*)ws;
    int* perm      = hdr + 16;
    int* tile_e    = perm + MAX_LEN;
    int* tile_m0   = tile_e + MAX_TILES;
    int* tile_rows = tile_m0 + MAX_TILES;
    size_t int_end = (size_t)(16 + MAX_LEN + 3 * MAX_TILES) * sizeof(int);
    size_t xp_off  = (int_end + 255) & ~(size_t)255;
    unsigned short* xp = (unsigned short*)(ws + xp_off);
    unsigned short* hb = xp + (size_t)MAX_LEN * DMODEL;

    (void)hipMemsetAsync(d_out, 0, (size_t)T_TOK * DMODEL * sizeof(float), stream);
    setup_kernel<<<1, 256, 0, stream>>>(counts, hdr, perm, tile_e, tile_m0, tile_rows);
    gather_kernel<<<(MAX_LEN * (DMODEL / 8)) / 256, 256, 0, stream>>>(x, perm, xp);
    gemm13_kernel<<<dim3(MAX_TILES, HDIM / 64), 256, 0, stream>>>(
        xp, w1, w3, hb, hdr, tile_e, tile_m0, tile_rows);
    gemm2_kernel<<<dim3(MAX_TILES, DMODEL / 64), 256, 0, stream>>>(
        hb, w2, perm, out, hdr, tile_e, tile_m0, tile_rows);
}
